// GIPAConv_33217277067275
// MI455X (gfx1250) — matrix operand order repaired and checked by reading, not compiled, not run on hardware
//
#include <hip/hip_runtime.h>

#define N_NODES   50000
#define N_EDGES   800000
#define NODE_F    256
#define EDGE_F    8
#define OUT_F     64
#define NEG_SLOPE 0.2f
#define EPSF      1e-9f

typedef __attribute__((ext_vector_type(16))) __bf16        v16bf;
typedef __attribute__((ext_vector_type(8)))  __bf16        v8bf;
typedef __attribute__((ext_vector_type(8)))  float         v8f;
typedef __attribute__((ext_vector_type(4)))  unsigned int  v4u;
typedef __attribute__((ext_vector_type(4)))  int           v4i;
typedef __attribute__((ext_vector_type(8)))  int           v8i;

#if __has_builtin(__builtin_amdgcn_tensor_load_to_lds)
#define HAVE_TDM_BUILTIN 1
#pragma message("probe: tensor_load_to_lds builtin AVAILABLE")
#else
#define HAVE_TDM_BUILTIN 0
#pragma message("probe: tensor_load_to_lds builtin NOT available -> inline asm path")
#endif

// LDS A-tile row stride in bf16 elements: 256 data + 8 pad (TDM pad_interval =
// 64x8B = one 256-elem row, pad_amount = 4 DW = 8 bf16). Keeps 16B row
// alignment and staggers banks (132-DW stride -> +4 banks per row).
#define ATILE_STRIDE 264

__device__ __forceinline__ float lrelu(float x) { return x > 0.f ? x : NEG_SLOPE * x; }

// Correct float atomic-max for mixed-sign values:
// non-negative floats order like signed ints; negative floats order reversed as unsigned.
__device__ __forceinline__ void atomic_max_f32(float* addr, float val) {
    if (val >= 0.f) atomicMax((int*)addr, __float_as_int(val));
    else            atomicMin((unsigned int*)addr, __float_as_uint(val));
}

__device__ __forceinline__ v16bf cat16(v8bf lo, v8bf hi) {
    return __builtin_shufflevector(lo, hi, 0, 1, 2, 3, 4, 5, 6, 7,
                                   8, 9, 10, 11, 12, 13, 14, 15);
}

// Issue one 2-D TDM load (group2/3 = NULL) and wait for completion.
__device__ __forceinline__ void tdm_load_2d(v4u g0, v8i g1) {
#if HAVE_TDM_BUILTIN
    v4i g2 = { 0, 0, 0, 0 };
    v4i g3 = { 0, 0, 0, 0 };
#if defined(__clang_major__) && (__clang_major__ >= 23)
    v8i g4 = { 0, 0, 0, 0, 0, 0, 0, 0 };
    __builtin_amdgcn_tensor_load_to_lds(g0, g1, g2, g3, g4, 0);
#else
    __builtin_amdgcn_tensor_load_to_lds(g0, g1, g2, g3, 0);
#endif
#else
    // Raw CDNA5 ISA: VADDR0 = 4-SGPR D# group0, VADDR1 = 8-SGPR D# group1,
    // VADDR2/VADDR3 omitted (NULL) -> tensors up to 2D.
    asm volatile("tensor_load_to_lds %0, %1" :: "s"(g0), "s"(g1) : "memory");
#endif
    __builtin_amdgcn_s_wait_tensorcnt(0);
}

// ---------------------------------------------------------------------------
// K0: initialize workspace accumulators (harness poisons ws with 0xAA).
// ---------------------------------------------------------------------------
__global__ void init_ws(float* __restrict__ m_dst, float* __restrict__ m_src,
                        float* __restrict__ s_dst, float* __restrict__ s_src,
                        float* __restrict__ msg) {
    size_t i = (size_t)blockIdx.x * blockDim.x + threadIdx.x;
    if (i < (size_t)N_NODES * OUT_F) {
        m_dst[i] = -3.0e38f;
        m_src[i] = -3.0e38f;
        s_dst[i] = 0.f;
        s_src[i] = 0.f;
        msg[i]   = 0.f;
    }
}

// K0b: one-time f32 -> bf16 conversion of node features (halves GEMM HBM read).
__global__ void conv_a16(const float* __restrict__ A, __bf16* __restrict__ A16) {
    int i = blockIdx.x * 256 + threadIdx.x;
    if (i < N_NODES * NODE_F) A16[i] = (__bf16)A[i];
}

// K0c: pack the four 64x256 weight matrices into one K-major bf16 block:
//      WcatT[k][col], col = wv*64 + r, so the 16 lanes of a B load read adjacent cols.
__global__ void pack_w16(const float* __restrict__ Wa_src, const float* __restrict__ Wa_dst,
                         const float* __restrict__ W_srcW, const float* __restrict__ W_dstW,
                         __bf16* __restrict__ WcatT) {
    int i = blockIdx.x * 256 + threadIdx.x;        // < 256*256 = 65536
    if (i >= NODE_F * 4 * OUT_F) return;
    int k = i >> 8, col = i & 255;
    int wv = col >> 6, r = col & 63;
    const float* W = (wv == 0) ? Wa_src : (wv == 1) ? Wa_dst : (wv == 2) ? W_srcW : W_dstW;
    WcatT[i] = (__bf16)W[(size_t)r * NODE_F + k];
}

// K0d: W_agg (64x64) -> K-major bf16: WaggT[k][n] = W_agg[n][k].
__global__ void pack_wagg(const float* __restrict__ W_agg, __bf16* __restrict__ WaggT) {
    int i = blockIdx.x * 256 + threadIdx.x;        // < 4096
    if (i >= OUT_F * OUT_F) return;
    int k = i >> 6, n = i & 63;
    WaggT[i] = (__bf16)W_agg[(size_t)n * OUT_F + k];
}

// ---------------------------------------------------------------------------
// K1: fused node projections via WMMA bf16 (f32 accumulate).
//   Block = 4 waves = one 16-row node tile. TDM DMAs the 16x256 bf16 A-tile
//   into LDS (tensor_load_to_lds, TENSORcnt); A fragments = two ds b128 loads,
//   B fragments = 16 strided global loads from K-major weights; wave w computes
//   the 64 output cols of weight matrix w (4 v_wmma / k-step reusing one A
//   fragment).
// ---------------------------------------------------------------------------
__global__ void proj_wmma(const __bf16* __restrict__ A16,    // [N, 256] bf16
                          const __bf16* __restrict__ WcatT,  // [256 k][256 col] bf16
                          const float*  __restrict__ b_dst,  // [64]
                          float* __restrict__ t_asrc, float* __restrict__ t_adst,
                          float* __restrict__ t_fc,   float* __restrict__ t_res) {
    __shared__ __align__(16) __bf16 lds_a16[16 * ATILE_STRIDE];   // 8.25 KB

    const int lane = threadIdx.x & 31;             // wave32 lane
    const int wv   = threadIdx.x >> 5;             // wave 0..3 = weight matrix id
    const int hh   = lane >> 4;                    // half-wave
    const int nl   = lane & 15;
    const int m0   = blockIdx.x * 16;              // node tile (50000 = 3125*16)

    if (threadIdx.x < 32) {
        // --- Tensor DMA descriptor (ISA 8.3-8.6), 2-D tensor, 2-byte elems ---
        unsigned long long ga = (unsigned long long)(__SIZE_TYPE__)(A16 + (size_t)m0 * NODE_F);
        unsigned int lds_off  = (unsigned int)(__SIZE_TYPE__)(&lds_a16[0]); // low 32b = LDS offset
        // group0: count=1 | lds_addr | global_addr[56:0] | type=2 ("image")
        v4u g0 = { 1u,
                   lds_off,
                   (unsigned int)ga,
                   ((unsigned int)(ga >> 32) & 0x01FFFFFFu) | (2u << 30) };
        // group1:
        //  w0: data_size=1(2B)<<16 | pad_enable<<20 | pad_interval=6(128DW=1row)<<22
        //      | pad_amount=3(4DW=8 bf16)<<25           = 0x07910000
        //  w1: tensor_dim0=256 in [31:16];  w2: tensor_dim1=50000(0xC350) in [31:16]
        //  w3: tile_dim0=256 in [31:16];    w4: tile_dim1=16;  w5: dim0_stride=256
        v8i g1 = { (int)0x07910000, (int)0x01000000, (int)0xC3500000, (int)0x01000000,
                   16, 256, 0, 0 };
        tdm_load_2d(g0, g1);
    }
    __syncthreads();

    float* T = (wv == 0) ? t_asrc : (wv == 1) ? t_adst : (wv == 2) ? t_fc : t_res;

    v8f c[4] = {};
    const __bf16* ar = &lds_a16[nl * ATILE_STRIDE];
    for (int kb = 0; kb < NODE_F; kb += 32) {
        // A fragment: lane = M row (both halves), K split per ISA 16-bit A layout.
        v16bf a = cat16(*(const v8bf*)&ar[kb + hh * 8],
                        *(const v8bf*)&ar[kb + 16 + hh * 8]);
        // B fragment: lane = N column, element = K in the same split order as A.
        const __bf16* br = WcatT + (size_t)kb * (4 * OUT_F) + wv * OUT_F + nl;
#pragma unroll
        for (int i = 0; i < 4; ++i) {
            v16bf b;
#pragma unroll
            for (int s = 0; s < 16; ++s)
                b[s] = br[(size_t)((s < 8) ? (hh * 8 + s) : (16 + hh * 8 + (s - 8))) * (4 * OUT_F) + i * 16];
            c[i] = __builtin_amdgcn_wmma_f32_16x16x32_bf16(false, a, false, b,
                                                           (short)0, c[i], false, false);
        }
    }

#pragma unroll
    for (int i = 0; i < 4; ++i) {
        float bias = (wv == 3) ? b_dst[i * 16 + nl] : 0.f;
#pragma unroll
        for (int v = 0; v < 8; ++v) {
            int row = m0 + hh * 8 + v;             // C/D layout: M = 8*half + vgpr
            T[(size_t)row * OUT_F + i * 16 + nl] = c[i][v] + bias;
        }
    }
}

// ---------------------------------------------------------------------------
// Edge logit: gather two L2-resident node rows + recompute 8->64 edge matvec.
// Recomputing (512 FMA) is far cheaper than streaming a 205 MB e[E,64] buffer.
// ---------------------------------------------------------------------------
__device__ __forceinline__ float edge_logit(int e, int f, int s, int d,
                                            const float* __restrict__ t_asrc,
                                            const float* __restrict__ t_adst,
                                            const float* __restrict__ feat_edge,
                                            const float* __restrict__ W_attn_edge) {
    float acc = t_asrc[(size_t)s * OUT_F + f] + t_adst[(size_t)d * OUT_F + f];
    const float* fe = feat_edge + (size_t)e * EDGE_F;
    const float* wr = W_attn_edge + (size_t)f * EDGE_F;
#pragma unroll
    for (int k = 0; k < EDGE_F; ++k) acc = fmaf(fe[k], wr[k], acc);
    return lrelu(acc);
}

// K2: segment max (per-dst and per-src), 64 channels.
__global__ void edge_max(const float* __restrict__ t_asrc, const float* __restrict__ t_adst,
                         const float* __restrict__ feat_edge, const float* __restrict__ W_attn_edge,
                         const int* __restrict__ src_idx, const int* __restrict__ dst_idx,
                         float* __restrict__ m_dst, float* __restrict__ m_src) {
    int tid = blockIdx.x * 256 + threadIdx.x;
    int e = tid >> 6, f = tid & 63;
    if (e >= N_EDGES) return;
    if (f == 0) {   // speculative prefetch of a future edge-feature line (dropped if OOB)
        __builtin_prefetch(feat_edge + (size_t)(e + 4096) * EDGE_F, 0, 0);
    }
    int s = src_idx[e], d = dst_idx[e];
    float v = edge_logit(e, f, s, d, t_asrc, t_adst, feat_edge, W_attn_edge);
    atomic_max_f32(&m_dst[(size_t)d * OUT_F + f], v);
    atomic_max_f32(&m_src[(size_t)s * OUT_F + f], v);
}

// K3: exp-sum accumulation for both softmaxes.
__global__ void edge_expsum(const float* __restrict__ t_asrc, const float* __restrict__ t_adst,
                            const float* __restrict__ feat_edge, const float* __restrict__ W_attn_edge,
                            const int* __restrict__ src_idx, const int* __restrict__ dst_idx,
                            const float* __restrict__ m_dst, const float* __restrict__ m_src,
                            float* __restrict__ s_dst, float* __restrict__ s_src) {
    int tid = blockIdx.x * 256 + threadIdx.x;
    int e = tid >> 6, f = tid & 63;
    if (e >= N_EDGES) return;
    int s = src_idx[e], d = dst_idx[e];
    float v = edge_logit(e, f, s, d, t_asrc, t_adst, feat_edge, W_attn_edge);
    size_t di = (size_t)d * OUT_F + f, si = (size_t)s * OUT_F + f;
    atomicAdd(&s_dst[di], __expf(v - m_dst[di]));
    atomicAdd(&s_src[si], __expf(v - m_src[si]));
}

// K4: a = sqrt(clip(softmax_dst)*clip(softmax_src)); msg[dst] += fc_src[src]*a.
__global__ void edge_msg(const float* __restrict__ t_asrc, const float* __restrict__ t_adst,
                         const float* __restrict__ feat_edge, const float* __restrict__ W_attn_edge,
                         const int* __restrict__ src_idx, const int* __restrict__ dst_idx,
                         const float* __restrict__ m_dst, const float* __restrict__ m_src,
                         const float* __restrict__ s_dst, const float* __restrict__ s_src,
                         const float* __restrict__ t_fc, float* __restrict__ msg) {
    int tid = blockIdx.x * 256 + threadIdx.x;
    int e = tid >> 6, f = tid & 63;
    if (e >= N_EDGES) return;
    int s = src_idx[e], d = dst_idx[e];
    float v = edge_logit(e, f, s, d, t_asrc, t_adst, feat_edge, W_attn_edge);
    size_t di = (size_t)d * OUT_F + f, si = (size_t)s * OUT_F + f;
    float ad = fmaxf(__expf(v - m_dst[di]) / s_dst[di], EPSF);
    float as = fmaxf(__expf(v - m_src[si]) / s_src[si], EPSF);
    float a  = sqrtf(ad * as);
    atomicAdd(&msg[di], t_fc[si] * a);
}

// ---------------------------------------------------------------------------
// K5: feature-dim norm + W_agg GEMM (WMMA) + bias + residual.
//     One wave handles 16 nodes; h staged in LDS as bf16 (row stride 72 keeps
//     16B alignment), then 4 col-tiles x 2 k-steps of b128 A loads, strided B loads + wmma.
// ---------------------------------------------------------------------------
#define HTILE_STRIDE 72
__global__ void norm_agg_wmma(const float* __restrict__ msg, const float* __restrict__ resid,
                              const float* __restrict__ scale, const float* __restrict__ offset,
                              const __bf16* __restrict__ WaggT,  // [64 k][64 n] bf16
                              const float* __restrict__ b_agg,
                              float* __restrict__ out) {
    __shared__ __align__(16) __bf16 hl16[16 * HTILE_STRIDE];
    const int lane = threadIdx.x;
    const int nb   = blockIdx.x * 16;

    if (lane < 16) {
        const float* r = msg + (size_t)(nb + lane) * OUT_F;
        float mean = 0.f;
        for (int j = 0; j < OUT_F; ++j) mean += r[j];
        mean *= (1.f / OUT_F);
        float var = 0.f;
        for (int j = 0; j < OUT_F; ++j) { float dd = r[j] - mean; var += dd * dd; }
        var = var * (1.f / OUT_F) + EPSF;
        float rs = rsqrtf(var);
        for (int j = 0; j < OUT_F; ++j)
            hl16[lane * HTILE_STRIDE + j] = (__bf16)((r[j] - mean) * scale[j] * rs + offset[j]);
    }
    __syncthreads();

    const int hh = lane >> 4, nl = lane & 15;
    const __bf16* ar = &hl16[nl * HTILE_STRIDE];
    for (int ct = 0; ct < 4; ++ct) {
        const int c0 = ct * 16;
        v8f c = {};
#pragma unroll
        for (int kb = 0; kb < OUT_F; kb += 32) {
            v16bf a = cat16(*(const v8bf*)&ar[kb + hh * 8],
                            *(const v8bf*)&ar[kb + 16 + hh * 8]);
            const __bf16* br = WaggT + (size_t)kb * OUT_F + c0 + nl;
            v16bf b;
#pragma unroll
            for (int s = 0; s < 16; ++s)
                b[s] = br[(size_t)((s < 8) ? (hh * 8 + s) : (16 + hh * 8 + (s - 8))) * OUT_F];
            c = __builtin_amdgcn_wmma_f32_16x16x32_bf16(false, a, false, b,
                                                        (short)0, c, false, false);
        }
#pragma unroll
        for (int v = 0; v < 8; ++v) {
            int node = nb + hh * 8 + v;
            int col  = c0 + nl;
            out[(size_t)node * OUT_F + col] =
                c[v] + b_agg[col] + resid[(size_t)node * OUT_F + col];
        }
    }
}

// ---------------------------------------------------------------------------
extern "C" void kernel_launch(void* const* d_in, const int* in_sizes, int n_in,
                              void* d_out, int out_size, void* d_ws, size_t ws_size,
                              hipStream_t stream) {
    (void)in_sizes; (void)n_in; (void)out_size; (void)ws_size;

    const float* feat_src    = (const float*)d_in[0];
    const float* feat_edge   = (const float*)d_in[1];
    const int*   src_idx     = (const int*)d_in[2];
    const int*   dst_idx     = (const int*)d_in[3];
    const float* W_src       = (const float*)d_in[4];
    const float* W_dst       = (const float*)d_in[5];
    const float* b_dst       = (const float*)d_in[6];
    const float* W_attn_src  = (const float*)d_in[7];
    const float* W_attn_dst  = (const float*)d_in[8];
    const float* W_attn_edge = (const float*)d_in[9];
    const float* scale       = (const float*)d_in[10];
    const float* offset      = (const float*)d_in[11];
    const float* W_agg       = (const float*)d_in[12];
    const float* b_agg       = (const float*)d_in[13];
    float*       out         = (float*)d_out;

    const size_t NF = (size_t)N_NODES * OUT_F;   // 3.2M floats = 12.8 MB per table
    float* ws     = (float*)d_ws;
    float* t_asrc = ws;  ws += NF;
    float* t_adst = ws;  ws += NF;
    float* t_fc   = ws;  ws += NF;
    float* t_res  = ws;  ws += NF;
    float* m_dst  = ws;  ws += NF;
    float* m_src  = ws;  ws += NF;
    float* s_dst  = ws;  ws += NF;
    float* s_src  = ws;  ws += NF;
    float* msg    = ws;  ws += NF;
    __bf16* A16   = (__bf16*)ws;  ws += (size_t)N_NODES * NODE_F / 2;  // 25.6 MB
    __bf16* WcatT = (__bf16*)ws;  ws += (size_t)NODE_F * 4 * OUT_F / 2;
    __bf16* WaggT = (__bf16*)ws;  ws += (size_t)OUT_F * OUT_F / 2;

    init_ws<<<(int)((NF + 255) / 256), 256, 0, stream>>>(m_dst, m_src, s_dst, s_src, msg);
    conv_a16<<<(N_NODES * NODE_F) / 256, 256, 0, stream>>>(feat_src, A16);
    pack_w16<<<(NODE_F * 4 * OUT_F) / 256, 256, 0, stream>>>(W_attn_src, W_attn_dst,
                                                             W_src, W_dst, WcatT);
    pack_wagg<<<(OUT_F * OUT_F + 255) / 256, 256, 0, stream>>>(W_agg, WaggT);

    proj_wmma<<<N_NODES / 16, 128, 0, stream>>>(A16, WcatT, b_dst,
                                                t_asrc, t_adst, t_fc, t_res);

    const int eb = (N_EDGES * 64) / 256;   // 200000 blocks, 1 thread per (edge,channel)
    edge_max<<<eb, 256, 0, stream>>>(t_asrc, t_adst, feat_edge, W_attn_edge,
                                     src_idx, dst_idx, m_dst, m_src);
    edge_expsum<<<eb, 256, 0, stream>>>(t_asrc, t_adst, feat_edge, W_attn_edge,
                                        src_idx, dst_idx, m_dst, m_src, s_dst, s_src);
    edge_msg<<<eb, 256, 0, stream>>>(t_asrc, t_adst, feat_edge, W_attn_edge,
                                     src_idx, dst_idx, m_dst, m_src, s_dst, s_src,
                                     t_fc, msg);

    norm_agg_wmma<<<N_NODES / 16, 32, 0, stream>>>(msg, t_res, scale, offset,
                                                   WaggT, b_agg, out);
}
